// MGAN_68496138436848
// MI455X (gfx1250) — matrix operand order repaired and checked by reading, not compiled, not run on hardware
//
#include <hip/hip_runtime.h>
#include <hip/hip_bf16.h>

// ---------------- problem constants ----------------
#define BB   512
#define LCc  128
#define LAa  16
#define EE   300
#define HH   300
#define H2   600      // 2*H (BiLSTM output)
#define NG   1200     // 4*H gate width
#define NGP  1216     // padded to multiple of 64
#define KE   320      // E (=H) padded to multiple of 32
#define KC   640      // concat K: [x(320) | h(320)]

typedef __attribute__((ext_vector_type(16))) __bf16 v16bf;
typedef __attribute__((ext_vector_type(8)))  float  v8f;

struct AF { union { v16bf v; uint4 q[2]; }; };

__device__ __forceinline__ float sigf(float x) { return 1.f / (1.f + __expf(-x)); }

__device__ __forceinline__ unsigned short f2bf(float f) {
    unsigned int u = __float_as_uint(f);
    u += 0x7fffu + ((u >> 16) & 1u);          // round-to-nearest-even
    return (unsigned short)(u >> 16);
}

// ---------------- small prep kernels ----------------
__global__ void k_lengths(const int* __restrict__ traw, const int* __restrict__ aspi,
                          const int* __restrict__ tleft,
                          int* clen, int* alen, int* llen) {
    int b = blockIdx.x * blockDim.x + threadIdx.x;
    if (b >= BB) return;
    int c = 0, a = 0, l = 0;
    for (int t = 0; t < LCc; ++t) c += (traw[b * LCc + t] != 0);
    for (int t = 0; t < LAa; ++t) a += (aspi[b * LAa + t] != 0);
    for (int t = 0; t < LCc; ++t) l += (tleft[b * LCc + t] != 0);
    clen[b] = c; alen[b] = a; llen[b] = l;
}

__global__ void k_ridx(const int* __restrict__ len, int* __restrict__ ridx, int T) {
    int i = blockIdx.x * blockDim.x + threadIdx.x;
    if (i >= BB * T) return;
    int b = i / T, t = i % T;
    int L = len[b];
    ridx[i] = (t < L) ? (L - 1 - t) : t;
}

// gather embedding rows -> bf16, K padded to KE with zeros
__global__ void k_gather(const int* __restrict__ tok, const float* __restrict__ embed,
                         unsigned short* __restrict__ dst, long total) {
    long i = (long)blockIdx.x * blockDim.x + threadIdx.x;
    if (i >= total) return;
    int k = (int)(i % KE);
    long row = i / KE;
    float v = 0.f;
    if (k < EE) { int tk = tok[row]; v = embed[(long)tk * EE + k]; }
    dst[i] = f2bf(v);
}

// build W_cat[n, 0:300]=W_ih[n], W_cat[n, 320:620]=W_hh[n], zero pad; NGP x KC bf16
__global__ void k_wcat(const float* __restrict__ Wih, const float* __restrict__ Whh,
                       unsigned short* __restrict__ dst) {
    long i = (long)blockIdx.x * blockDim.x + threadIdx.x;
    if (i >= (long)NGP * KC) return;
    int k = (int)(i % KC);
    int n = (int)(i / KC);
    float v = 0.f;
    if (n < NG) {
        if (k < EE)                    v = Wih[(long)n * EE + k];
        else if (k >= KE && k < KE + HH) v = Whh[(long)n * HH + (k - KE)];
    }
    dst[i] = f2bf(v);
}

__global__ void k_reset(unsigned short* __restrict__ hb, float* __restrict__ c) {
    int i = blockIdx.x * blockDim.x + threadIdx.x;
    if (i < BB * KE) hb[i] = 0;
    if (i < BB * HH) c[i] = 0.f;
}

// ---------------- WMMA LSTM step GEMM ----------------
// g[b, 0:1200] = [x_{b,tsrc} | h_b] @ Wcat^T + bias     (bf16 x bf16 -> f32 acc)
__device__ __forceinline__ void wmma_mac(const unsigned short* __restrict__ arow, int ak,
                                         const unsigned short* __restrict__ W, int wk,
                                         int colBase, int lane, v8f* acc) {
    const int half = lane >> 4;
    AF a;
    // A 16x32 bf16 lane layout: lanes0-15 hold K {0..7,16..23}, lanes16-31 hold K {8..15,24..31}
    a.q[0] = *(const uint4*)(arow + ak + half * 8);
    a.q[1] = *(const uint4*)(arow + ak + half * 8 + 16);
#pragma unroll
    for (int nt = 0; nt < 4; ++nt) {
        const int n = colBase + nt * 16 + (lane & 15);
        // B 32x16 bf16 lane layout: lanes0-15 K {0..7,16..23}, lanes16-31 K {8..15,24..31} (matches A)
        const unsigned short* wrow = W + (size_t)n * KC + wk + half * 8;
        AF bm;
        bm.q[0] = *(const uint4*)(wrow + 0);
        bm.q[1] = *(const uint4*)(wrow + 16);
        acc[nt] = __builtin_amdgcn_wmma_f32_16x16x32_bf16(
            false, a.v, false, bm.v, (short)0, acc[nt], false, false);
    }
}

__launch_bounds__(128)
__global__ void k_step_gemm(const unsigned short* __restrict__ xb,   // [BB*T, KE]
                            const unsigned short* __restrict__ hb,   // [BB, KE]
                            const unsigned short* __restrict__ Wc,   // [NGP, KC]
                            const float* __restrict__ bias,          // [NG]
                            float* __restrict__ g,                   // [BB, NGP]
                            const int* __restrict__ ridx,
                            int T, int step, int useR) {
    const int lane = threadIdx.x & 31;
    const int wave = threadIdx.x >> 5;               // 4 waves / block
    const int half = lane >> 4;
    const int mBase = blockIdx.x * 64 + wave * 16;   // 16 rows per wave
    const int colBase = blockIdx.y * 64;             // 64 cols per wave (4 tiles)
    const int mRow = mBase + (lane & 15);            // row == batch index
    const int tsrc = useR ? ridx[mRow * T + step] : step;

    const unsigned short* xrow = xb + ((size_t)mRow * T + tsrc) * KE;
    const unsigned short* hrow = hb + (size_t)mRow * KE;

    v8f acc[4];
#pragma unroll
    for (int nt = 0; nt < 4; ++nt)
#pragma unroll
        for (int r = 0; r < 8; ++r) acc[nt][r] = 0.f;

    // K part 1: x_t against Wcat cols [0, KE)
    for (int kb = 0; kb < KE; kb += 32) wmma_mac(xrow, kb, Wc, kb, colBase, lane, acc);
    // K part 2: h_{t-1} against Wcat cols [KE, 2*KE)
    for (int kb = 0; kb < KE; kb += 32) wmma_mac(hrow, kb, Wc, KE + kb, colBase, lane, acc);

#pragma unroll
    for (int nt = 0; nt < 4; ++nt) {
        const int n = colBase + nt * 16 + (lane & 15);
        if (n < NG) {
            const float bv = bias[n];
            const int m0 = mBase + 8 * half;
#pragma unroll
            for (int r = 0; r < 8; ++r)
                g[(size_t)(m0 + r) * NGP + n] = acc[nt][r] + bv;
        }
    }
}

// gate nonlinearity + state update + output scatter
__global__ void k_gate(const float* __restrict__ g, float* __restrict__ c,
                       unsigned short* __restrict__ hb, float* __restrict__ out,
                       const int* __restrict__ ridx, int T, int step, int useR, int off) {
    int i = blockIdx.x * blockDim.x + threadIdx.x;
    if (i >= BB * HH) return;
    int b = i / HH, j = i % HH;
    const float* gr = g + (size_t)b * NGP;
    float iv = gr[j], fv = gr[HH + j], gv = gr[2 * HH + j], ov = gr[3 * HH + j];
    float cs = sigf(fv) * c[i] + sigf(iv) * tanhf(gv);
    c[i] = cs;
    float h = sigf(ov) * tanhf(cs);
    hb[(size_t)b * KE + j] = f2bf(h);
    int td = useR ? ridx[b * T + step] : step;   // ridx is an involution
    out[((size_t)b * T + td) * H2 + off + j] = h;
}

// ---------------- post-LSTM scalar kernels ----------------
__global__ void k_ctxw(float* __restrict__ out, const int* clen, const int* alen,
                       const int* llen) {
    int i = blockIdx.x * blockDim.x + threadIdx.x;
    if (i >= BB * LCc) return;
    int b = i / LCc, t = i % LCc;
    float w;
    if (t >= clen[b]) w = 0.f;                       // sequence mask
    else {
        float s = (float)llen[b];
        float e = s + (float)alen[b] - 1.f;
        float sl = (float)(LCc - alen[b]);
        float j = (float)t;
        if (j < s)       w = 1.f - (s - j) / sl;
        else if (j <= e) w = 0.f;
        else             w = 1.f - (j - e) / sl;
    }
    float* p = out + (size_t)i * H2;
    for (int d = 0; d < H2; ++d) p[d] *= w;
}

__global__ void k_aspmask(float* __restrict__ out, const int* alen) {
    int i = blockIdx.x * blockDim.x + threadIdx.x;
    if (i >= BB * LAa) return;
    int b = i / LAa, t = i % LAa;
    if (t >= alen[b]) {
        float* p = out + (size_t)i * H2;
        for (int d = 0; d < H2; ++d) p[d] = 0.f;
    }
}

__global__ void k_pool(const float* __restrict__ rows, const int* __restrict__ len,
                       float* __restrict__ pool, int T) {
    int i = blockIdx.x * blockDim.x + threadIdx.x;
    if (i >= BB * H2) return;
    int b = i / H2, d = i % H2;
    const float* p = rows + (size_t)b * T * H2 + d;
    float s = 0.f;
    for (int t = 0; t < T; ++t) s += p[(size_t)t * H2];
    pool[i] = s / (float)len[b];
}

__global__ void k_rowdot(const float* __restrict__ rows, const float* __restrict__ vec,
                         float* __restrict__ out, int total) {
    int i = blockIdx.x * blockDim.x + threadIdx.x;
    if (i >= total) return;
    const float* r = rows + (size_t)i * H2;
    float s = 0.f;
    for (int d = 0; d < H2; ++d) s += r[d] * vec[d];
    out[i] = s;
}

__global__ void k_align(const float* __restrict__ ctx, const float* __restrict__ asp,
                        const float* __restrict__ w3, const float* __restrict__ cw1,
                        const float* __restrict__ aw2, float* __restrict__ al) {
    int i = blockIdx.x * blockDim.x + threadIdx.x;
    if (i >= BB * LCc * LAa) return;
    int j = i % LAa;
    int bi = i / LAa;             // b*LCc + t
    int b = bi / LCc;
    const float* cr = ctx + (size_t)bi * H2;
    const float* ar = asp + ((size_t)b * LAa + j) * H2;
    float s = 0.f;
    for (int d = 0; d < H2; ++d) s += cr[d] * ar[d] * w3[d];
    al[i] = s + cw1[bi] + aw2[b * LAa + j];
}

__global__ void k_maxj(const float* __restrict__ al, float* __restrict__ m) {
    int i = blockIdx.x * blockDim.x + threadIdx.x;
    if (i >= BB * LCc) return;
    const float* a = al + (size_t)i * LAa;
    float mm = a[0];
    for (int j = 1; j < LAa; ++j) mm = fmaxf(mm, a[j]);
    m[i] = mm;
}

__global__ void k_maxi(const float* __restrict__ al, float* __restrict__ m) {
    int i = blockIdx.x * blockDim.x + threadIdx.x;
    if (i >= BB * LAa) return;
    int b = i / LAa, j = i % LAa;
    float mm = al[(size_t)b * LCc * LAa + j];
    for (int t = 1; t < LCc; ++t)
        mm = fmaxf(mm, al[((size_t)b * LCc + t) * LAa + j]);
    m[i] = mm;
}

__global__ void k_softmax(const float* __restrict__ x, float* __restrict__ y, int L) {
    int b = blockIdx.x * blockDim.x + threadIdx.x;
    if (b >= BB) return;
    const float* r = x + (size_t)b * L;
    float* o = y + (size_t)b * L;
    float m = r[0];
    for (int i = 1; i < L; ++i) m = fmaxf(m, r[i]);
    float s = 0.f;
    for (int i = 0; i < L; ++i) { float e = __expf(r[i] - m); o[i] = e; s += e; }
    float inv = 1.f / s;
    for (int i = 0; i < L; ++i) o[i] *= inv;
}

__global__ void k_wsum(const float* __restrict__ rows, const float* __restrict__ w,
                       float* __restrict__ out, int T) {
    int i = blockIdx.x * blockDim.x + threadIdx.x;
    if (i >= BB * H2) return;
    int b = i / H2, d = i % H2;
    float s = 0.f;
    for (int t = 0; t < T; ++t)
        s += w[b * T + t] * rows[((size_t)b * T + t) * H2 + d];
    out[i] = s;
}

__global__ void k_matpool(const float* __restrict__ Wm, const float* __restrict__ pool,
                          float* __restrict__ u) {
    int i = blockIdx.x * blockDim.x + threadIdx.x;
    if (i >= BB * H2) return;
    int b = i / H2, d = i % H2;
    const float* wr = Wm + (size_t)d * H2;
    const float* p = pool + (size_t)b * H2;
    float s = 0.f;
    for (int e = 0; e < H2; ++e) s += wr[e] * p[e];
    u[i] = s;
}

__global__ void k_score(const float* __restrict__ rows, const float* __restrict__ u,
                        float* __restrict__ sc, int T) {
    int i = blockIdx.x * blockDim.x + threadIdx.x;
    if (i >= BB * T) return;
    int b = i / T;
    const float* r = rows + (size_t)i * H2;
    const float* ub = u + (size_t)b * H2;
    float s = 0.f;
    for (int d = 0; d < H2; ++d) s += r[d] * ub[d];
    sc[i] = s;
}

__global__ void k_final(const float* __restrict__ c1, const float* __restrict__ f1,
                        const float* __restrict__ f2, const float* __restrict__ c2,
                        const float* __restrict__ dW, const float* __restrict__ db,
                        float* __restrict__ out) {
    int i = blockIdx.x * blockDim.x + threadIdx.x;
    if (i >= BB * 3) return;
    int b = i / 3, p = i % 3;
    const float* w = dW + (size_t)p * 4 * H2;
    const float* A = c1 + (size_t)b * H2;
    const float* Bv = f1 + (size_t)b * H2;
    const float* C = f2 + (size_t)b * H2;
    const float* D = c2 + (size_t)b * H2;
    float s = db[p];
    for (int d = 0; d < H2; ++d) s += w[d]          * A[d];
    for (int d = 0; d < H2; ++d) s += w[H2 + d]     * Bv[d];
    for (int d = 0; d < H2; ++d) s += w[2 * H2 + d] * C[d];
    for (int d = 0; d < H2; ++d) s += w[3 * H2 + d] * D[d];
    out[i] = s;
}

// ---------------- host launcher ----------------
static inline size_t alignUp256(size_t x) { return (x + 255) & ~(size_t)255; }

extern "C" void kernel_launch(void* const* d_in, const int* in_sizes, int n_in,
                              void* d_out, int out_size, void* d_ws, size_t ws_size,
                              hipStream_t stream) {
    (void)in_sizes; (void)n_in; (void)out_size; (void)ws_size;

    const int*   traw  = (const int*)d_in[0];
    const int*   aspi  = (const int*)d_in[1];
    const int*   tleft = (const int*)d_in[2];
    const float* embed = (const float*)d_in[3];
    const float* cWif = (const float*)d_in[4],  *cWhf = (const float*)d_in[5],  *cbf = (const float*)d_in[6];
    const float* cWib = (const float*)d_in[7],  *cWhb = (const float*)d_in[8],  *cbb = (const float*)d_in[9];
    const float* aWif = (const float*)d_in[10], *aWhf = (const float*)d_in[11], *abf = (const float*)d_in[12];
    const float* aWib = (const float*)d_in[13], *aWhb = (const float*)d_in[14], *abb = (const float*)d_in[15];
    const float* w_u   = (const float*)d_in[16];   // [1800]
    const float* w_a2c = (const float*)d_in[17];   // [600,600]
    const float* w_c2a = (const float*)d_in[18];
    const float* dW    = (const float*)d_in[19];   // [3,2400]
    const float* db    = (const float*)d_in[20];
    float* outp = (float*)d_out;

    char* p = (char*)d_ws;
    auto take = [&](size_t bytes) { void* r = (void*)p; p += alignUp256(bytes); return r; };

    unsigned short* ctx_bf = (unsigned short*)take((size_t)BB * LCc * KE * 2);
    unsigned short* asp_bf = (unsigned short*)take((size_t)BB * LAa * KE * 2);
    unsigned short* Wc_cf  = (unsigned short*)take((size_t)NGP * KC * 2);
    unsigned short* Wc_cb  = (unsigned short*)take((size_t)NGP * KC * 2);
    unsigned short* Wc_af  = (unsigned short*)take((size_t)NGP * KC * 2);
    unsigned short* Wc_ab  = (unsigned short*)take((size_t)NGP * KC * 2);
    unsigned short* hb     = (unsigned short*)take((size_t)BB * KE * 2);
    float* cst     = (float*)take((size_t)BB * HH * 4);
    float* g       = (float*)take((size_t)BB * NGP * 4);
    float* ctx_out = (float*)take((size_t)BB * LCc * H2 * 4);
    float* asp_out = (float*)take((size_t)BB * LAa * H2 * 4);
    int* clen  = (int*)take(BB * 4);
    int* alen  = (int*)take(BB * 4);
    int* llen  = (int*)take(BB * 4);
    int* ridxC = (int*)take((size_t)BB * LCc * 4);
    int* ridxA = (int*)take((size_t)BB * LAa * 4);
    float* ctx_pool = (float*)take((size_t)BB * H2 * 4);
    float* asp_pool = (float*)take((size_t)BB * H2 * 4);
    float* cw1   = (float*)take((size_t)BB * LCc * 4);
    float* aw2   = (float*)take((size_t)BB * LAa * 4);
    float* alnb  = (float*)take((size_t)BB * LCc * LAa * 4);
    float* mrow  = (float*)take((size_t)BB * LCc * 4);
    float* mcol  = (float*)take((size_t)BB * LAa * 4);
    float* alpha = (float*)take((size_t)BB * LCc * 4);
    float* beta  = (float*)take((size_t)BB * LAa * 4);
    float* sc1   = (float*)take((size_t)BB * LCc * 4);
    float* a1v   = (float*)take((size_t)BB * LCc * 4);
    float* sc2   = (float*)take((size_t)BB * LAa * 4);
    float* a2v   = (float*)take((size_t)BB * LAa * 4);
    float* u1 = (float*)take((size_t)BB * H2 * 4);
    float* u2 = (float*)take((size_t)BB * H2 * 4);
    float* f1 = (float*)take((size_t)BB * H2 * 4);   // f_asp2ctx
    float* f2 = (float*)take((size_t)BB * H2 * 4);   // f_ctx2asp
    float* c1 = (float*)take((size_t)BB * H2 * 4);   // c_asp2ctx
    float* c2 = (float*)take((size_t)BB * H2 * 4);   // c_ctx2asp

    const int TB = 256;
    auto blocks = [&](long n) { return (unsigned)((n + TB - 1) / TB); };

    // prep
    k_lengths<<<blocks(BB), TB, 0, stream>>>(traw, aspi, tleft, clen, alen, llen);
    k_ridx<<<blocks((long)BB * LCc), TB, 0, stream>>>(clen, ridxC, LCc);
    k_ridx<<<blocks((long)BB * LAa), TB, 0, stream>>>(alen, ridxA, LAa);
    k_gather<<<blocks((long)BB * LCc * KE), TB, 0, stream>>>(traw, embed, ctx_bf, (long)BB * LCc * KE);
    k_gather<<<blocks((long)BB * LAa * KE), TB, 0, stream>>>(aspi, embed, asp_bf, (long)BB * LAa * KE);
    k_wcat<<<blocks((long)NGP * KC), TB, 0, stream>>>(cWif, cWhf, Wc_cf);
    k_wcat<<<blocks((long)NGP * KC), TB, 0, stream>>>(cWib, cWhb, Wc_cb);
    k_wcat<<<blocks((long)NGP * KC), TB, 0, stream>>>(aWif, aWhf, Wc_af);
    k_wcat<<<blocks((long)NGP * KC), TB, 0, stream>>>(aWib, aWhb, Wc_ab);

    // BiLSTM scans (graph-captured kernel chain; serial dependence on stream)
    dim3 gemmGrid(BB / 64, NGP / 64);   // (8, 19), 128 threads = 4 waves
    auto run_scan = [&](const unsigned short* xb, const unsigned short* Wc, const float* bias,
                        float* outBuf, int T, const int* ridx, int useR, int off) {
        k_reset<<<blocks((long)BB * KE), TB, 0, stream>>>(hb, cst);
        for (int s = 0; s < T; ++s) {
            k_step_gemm<<<gemmGrid, 128, 0, stream>>>(xb, hb, Wc, bias, g, ridx, T, s, useR);
            k_gate<<<blocks((long)BB * HH), TB, 0, stream>>>(g, cst, hb, outBuf, ridx, T, s, useR, off);
        }
    };
    run_scan(ctx_bf, Wc_cf, cbf, ctx_out, LCc, ridxC, 0, 0);
    run_scan(ctx_bf, Wc_cb, cbb, ctx_out, LCc, ridxC, 1, HH);
    run_scan(asp_bf, Wc_af, abf, asp_out, LAa, ridxA, 0, 0);
    run_scan(asp_bf, Wc_ab, abb, asp_out, LAa, ridxA, 1, HH);

    // position weighting / masking
    k_ctxw<<<blocks((long)BB * LCc), TB, 0, stream>>>(ctx_out, clen, alen, llen);
    k_aspmask<<<blocks((long)BB * LAa), TB, 0, stream>>>(asp_out, alen);

    // pooling
    k_pool<<<blocks((long)BB * H2), TB, 0, stream>>>(ctx_out, clen, ctx_pool, LCc);
    k_pool<<<blocks((long)BB * H2), TB, 0, stream>>>(asp_out, alen, asp_pool, LAa);

    // trilinear alignment
    k_rowdot<<<blocks((long)BB * LCc), TB, 0, stream>>>(ctx_out, w_u + 0,   cw1, BB * LCc);
    k_rowdot<<<blocks((long)BB * LAa), TB, 0, stream>>>(asp_out, w_u + 600, aw2, BB * LAa);
    k_align<<<blocks((long)BB * LCc * LAa), TB, 0, stream>>>(ctx_out, asp_out, w_u + 1200, cw1, aw2, alnb);

    // fine-grained attention
    k_maxj<<<blocks((long)BB * LCc), TB, 0, stream>>>(alnb, mrow);
    k_maxi<<<blocks((long)BB * LAa), TB, 0, stream>>>(alnb, mcol);
    k_softmax<<<blocks(BB), TB, 0, stream>>>(mrow, alpha, LCc);
    k_softmax<<<blocks(BB), TB, 0, stream>>>(mcol, beta, LAa);
    k_wsum<<<blocks((long)BB * H2), TB, 0, stream>>>(ctx_out, alpha, f1, LCc);
    k_wsum<<<blocks((long)BB * H2), TB, 0, stream>>>(asp_out, beta, f2, LAa);

    // coarse-grained attention
    k_matpool<<<blocks((long)BB * H2), TB, 0, stream>>>(w_a2c, asp_pool, u1);
    k_score<<<blocks((long)BB * LCc), TB, 0, stream>>>(ctx_out, u1, sc1, LCc);
    k_softmax<<<blocks(BB), TB, 0, stream>>>(sc1, a1v, LCc);
    k_wsum<<<blocks((long)BB * H2), TB, 0, stream>>>(ctx_out, a1v, c1, LCc);

    k_matpool<<<blocks((long)BB * H2), TB, 0, stream>>>(w_c2a, ctx_pool, u2);
    k_score<<<blocks((long)BB * LAa), TB, 0, stream>>>(asp_out, u2, sc2, LAa);
    k_softmax<<<blocks(BB), TB, 0, stream>>>(sc2, a2v, LAa);
    k_wsum<<<blocks((long)BB * H2), TB, 0, stream>>>(asp_out, a2v, c2, LAa);

    // classifier: feat = [c_asp2ctx, f_asp2ctx, f_ctx2asp, c_ctx2asp] @ dW^T + db
    k_final<<<blocks((long)BB * 3), TB, 0, stream>>>(c1, f1, f2, c2, dW, db, outp);
}
